// BertSelfAttention_61186104099215
// MI455X (gfx1250) — matrix operand order repaired and checked by reading, not compiled, not run on hardware
//
#include <hip/hip_runtime.h>

// ---------------------------------------------------------------------------
// BERT self-attention for gfx1250 (MI455X), bf16 WMMA pipeline.
//   B=2, S=2048, H=1024, NH=16, HD=64
// Kernel 1: fused QKV projection GEMM (f32 in -> bf16 Q/K/Vt in workspace)
// Kernel 2: flash-style attention, K/V tiles async-staged to LDS
// ---------------------------------------------------------------------------

#define HSZ   1024
#define NH    16
#define HD    64
#define BB    2
#define SS    2048
#define MROWS (BB * SS)          // 4096

typedef __bf16 v16bf __attribute__((ext_vector_type(16)));
typedef float  v8f   __attribute__((ext_vector_type(8)));
typedef int    v4i   __attribute__((ext_vector_type(4)));
typedef unsigned short u16;
typedef unsigned int   u32;

#define AS1 __attribute__((address_space(1)))
#define AS3 __attribute__((address_space(3)))

#if __has_builtin(__builtin_amdgcn_global_load_async_to_lds_b128) && \
    __has_builtin(__builtin_amdgcn_s_wait_asynccnt)
#define HAVE_ASYNC_LDS 1
#else
#define HAVE_ASYNC_LDS 0
#endif

// f32 -> bf16 via native conversion (RNE); packed pair -> one dword
__device__ __forceinline__ u32 pkbf(float a, float b) {
    union { __bf16 h[2]; u32 u; } p;
    p.h[0] = (__bf16)a;
    p.h[1] = (__bf16)b;
    return p.u;
}
__device__ __forceinline__ u16 bf1(float a) {
    union { __bf16 h; u16 u; } p;
    p.h = (__bf16)a;
    return p.u;
}

// 16 bf16 values (one WMMA A or B fragment) viewed as 2x 128-bit chunks
union FragB16 {
    v16bf h;
    uint4 q[2];
};

__device__ __forceinline__ v8f wmma_bf16(const FragB16& a, const FragB16& b, v8f c) {
    // D = A(16x32 bf16) * B(32x16 bf16) + C(16x16 f32)
    return __builtin_amdgcn_wmma_f32_16x16x32_bf16(
        /*neg_a=*/false, a.h, /*neg_b=*/false, b.h,
        /*c_mod=*/(short)0, c, /*reuse_a=*/false, /*reuse_b=*/false);
}

// ---------------------------------------------------------------------------
// Kernel 1: out = X @ W^T + b   for W in {Wq, Wk, Wv} (blockIdx.z selects)
//   X: [4096, 1024] f32,  W: [1024, 1024] f32 (row = output channel)
//   Q,K -> [b, h, s, d] bf16 ; V -> transposed [b, h, d, s] bf16
// Block tile 128x64, 256 threads = 8 waves, each wave a 32x32 patch.
// ---------------------------------------------------------------------------
#define K1_ASTR 40   // 32 + 8 halves padding
#define K1_WSTR 40

__global__ __launch_bounds__(256) void qkv_gemm_kernel(
    const float* __restrict__ X,
    const float* __restrict__ Wq, const float* __restrict__ bq,
    const float* __restrict__ Wk, const float* __restrict__ bk,
    const float* __restrict__ Wv, const float* __restrict__ bv,
    u16* __restrict__ Qo, u16* __restrict__ Ko, u16* __restrict__ Vt)
{
    __shared__ __align__(16) u16 a_lds[128 * K1_ASTR];
    __shared__ __align__(16) u16 w_lds[64 * K1_WSTR];

    const int tid  = threadIdx.x;
    const int wid  = tid >> 5;
    const int lane = tid & 31;
    const int half = lane >> 4;
    const int ln   = lane & 15;
    const int waveM = wid & 3;        // 4 row-waves (32 rows each)
    const int waveN = wid >> 2;       // 2 col-waves (32 cols each)

    const float* W;
    const float* bias;
    if (blockIdx.z == 0)      { W = Wq; bias = bq; }
    else if (blockIdx.z == 1) { W = Wk; bias = bk; }
    else                      { W = Wv; bias = bv; }

    const int rowBase = blockIdx.x * 128;
    const int colBase = blockIdx.y * 64;

    const v8f zero8 = {0.f, 0.f, 0.f, 0.f, 0.f, 0.f, 0.f, 0.f};
    v8f acc[2][2];
    acc[0][0] = zero8; acc[0][1] = zero8; acc[1][0] = zero8; acc[1][1] = zero8;

    #pragma unroll 1
    for (int k0 = 0; k0 < HSZ; k0 += 32) {
        // ---- stage A (128x32 f32 -> bf16 LDS), 4 float4 per thread ----
        #pragma unroll
        for (int c = 0; c < 4; ++c) {
            int cid  = tid + c * 256;         // 0..1023
            int row  = cid >> 3;
            int col4 = cid & 7;
            float4 f = *(const float4*)(X + (rowBase + row) * HSZ + k0 + col4 * 4);
            uint2 p;
            p.x = pkbf(f.x, f.y);
            p.y = pkbf(f.z, f.w);
            *(uint2*)(a_lds + row * K1_ASTR + col4 * 4) = p;
        }
        // ---- stage W (64x32 f32 -> bf16 LDS), 2 float4 per thread ----
        #pragma unroll
        for (int c = 0; c < 2; ++c) {
            int cid  = tid + c * 256;         // 0..511
            int row  = cid >> 3;
            int col4 = cid & 7;
            float4 f = *(const float4*)(W + (colBase + row) * HSZ + k0 + col4 * 4);
            uint2 p;
            p.x = pkbf(f.x, f.y);
            p.y = pkbf(f.z, f.w);
            *(uint2*)(w_lds + row * K1_WSTR + col4 * 4) = p;
        }
        __syncthreads();

        // ---- WMMA fragments straight out of LDS (all ds_load_b128) ----
        FragB16 afr[2], bfr[2];
        #pragma unroll
        for (int mi = 0; mi < 2; ++mi) {
            // A layout: lane-half 0 -> K 0..7 & 16..23, half 1 -> 8..15 & 24..31
            int idx = (waveM * 32 + mi * 16 + ln) * K1_ASTR + half * 8;
            afr[mi].q[0] = *(const uint4*)(a_lds + idx);
            afr[mi].q[1] = *(const uint4*)(a_lds + idx + 16);
        }
        #pragma unroll
        for (int ni = 0; ni < 2; ++ni) {
            // B layout: lane-half 0 -> K 0..7 & 16..23, half 1 -> 8..15 & 24..31
            int idx = (waveN * 32 + ni * 16 + ln) * K1_WSTR + half * 8;
            bfr[ni].q[0] = *(const uint4*)(w_lds + idx);
            bfr[ni].q[1] = *(const uint4*)(w_lds + idx + 16);
        }
        #pragma unroll
        for (int mi = 0; mi < 2; ++mi)
            #pragma unroll
            for (int ni = 0; ni < 2; ++ni)
                acc[mi][ni] = wmma_bf16(afr[mi], bfr[ni], acc[mi][ni]);

        __syncthreads();
    }

    // ---- epilogue: bias, convert, scatter to head-major layouts ----
    #pragma unroll
    for (int ni = 0; ni < 2; ++ni) {
        int nG   = colBase + waveN * 32 + ni * 16 + ln;   // output channel
        float bv_ = bias[nG];
        int hIdx = nG >> 6;       // / HD
        int d    = nG & 63;
        #pragma unroll
        for (int mi = 0; mi < 2; ++mi) {
            #pragma unroll
            for (int r = 0; r < 8; ++r) {
                int mG = rowBase + waveM * 32 + mi * 16 + r + half * 8; // token row
                int bI = mG >> 11;           // / SS
                int s  = mG & (SS - 1);
                u16 hv = bf1(acc[mi][ni][r] + bv_);
                if (blockIdx.z == 2) {
                    // V transposed: [b, h, d, s]
                    Vt[(((bI * NH + hIdx) * HD + d) * SS) + s] = hv;
                } else {
                    u16* dst = (blockIdx.z == 0) ? Qo : Ko;
                    dst[(((bI * NH + hIdx) * SS + s) * HD) + d] = hv;
                }
            }
        }
    }
}

// ---------------------------------------------------------------------------
// Kernel 2: flash-style attention with block-shared K/V tiles.
// grid = (S/128, NH, B); 256 threads = 8 waves; each wave owns 16 queries.
// Per 64-key tile: async-stage K (64x64) + V^T (64x64) bf16 tiles to LDS once
// per block, then 8 WMMAs for S = Q K^T, online softmax, 8 WMMAs for O += P V.
// ---------------------------------------------------------------------------
#define K2_STR 72   // 64 + 8 halves padding
#define P_STR  72

__global__ __launch_bounds__(256) void flash_attn_kernel(
    const u16* __restrict__ Q, const u16* __restrict__ K,
    const u16* __restrict__ Vt, const float* __restrict__ mask,
    float* __restrict__ out)
{
    __shared__ __align__(16) u16 k_lds[64 * K2_STR];    // [key][d]
    __shared__ __align__(16) u16 v_lds[64 * K2_STR];    // [d][key]
    __shared__ __align__(16) u16 p_lds[8][16 * P_STR];  // per-wave P staging

    const int tid  = threadIdx.x;
    const int wid  = tid >> 5;
    const int lane = tid & 31;
    const int half = lane >> 4;
    const int ln   = lane & 15;

    const int b    = blockIdx.z;
    const int head = blockIdx.y;
    const int q0   = blockIdx.x * 128 + wid * 16;

    const u16* Qh = Q  + ((b * NH + head) * SS) * HD;
    const u16* Kh = K  + ((b * NH + head) * SS) * HD;
    const u16* Vh = Vt + ((b * NH + head) * HD) * SS;
    const float* mk = mask + b * SS;

    // Q fragments resident in registers for the whole key loop (2 K-steps of 32)
    FragB16 qf[2];
    #pragma unroll
    for (int ks = 0; ks < 2; ++ks) {
        int idx = (q0 + ln) * HD + ks * 32 + half * 8;
        qf[ks].q[0] = *(const uint4*)(Qh + idx);
        qf[ks].q[1] = *(const uint4*)(Qh + idx + 16);
    }

    const v8f zero8 = {0.f, 0.f, 0.f, 0.f, 0.f, 0.f, 0.f, 0.f};
    v8f o_acc[4] = {zero8, zero8, zero8, zero8};    // 16 queries x 64 d (f32)
    float m_run[8], l_run[8];
    #pragma unroll
    for (int r = 0; r < 8; ++r) { m_run[r] = -1e30f; l_run[r] = 0.f; }

    u16* pw = &p_lds[wid][0];

    #pragma unroll 1
    for (int k0 = 0; k0 < SS; k0 += 64) {
        __syncthreads();   // previous tile's LDS reads complete before overwrite

        // ---- stage K tile [64 keys][64 d] and V^T tile [64 d][64 keys] ----
        // 64x64 bf16 = 512 x 16B chunks each; 2 chunks per thread per tile.
        #pragma unroll
        for (int c = 0; c < 2; ++c) {
            int cid  = tid + c * 256;       // 0..511
            int row  = cid >> 3;
            int col8 = cid & 7;
            const u16* gk = Kh + (k0 + row) * HD + col8 * 8;
            const u16* gv = Vh + row * SS + k0 + col8 * 8;
            u16* lk = k_lds + row * K2_STR + col8 * 8;
            u16* lv = v_lds + row * K2_STR + col8 * 8;
#if HAVE_ASYNC_LDS
            __builtin_amdgcn_global_load_async_to_lds_b128(
                (AS1 v4i*)(u16*)gk, (AS3 v4i*)lk, 0, 0);
            __builtin_amdgcn_global_load_async_to_lds_b128(
                (AS1 v4i*)(u16*)gv, (AS3 v4i*)lv, 0, 0);
#else
            *(uint4*)lk = *(const uint4*)gk;
            *(uint4*)lv = *(const uint4*)gv;
#endif
        }
#if HAVE_ASYNC_LDS
        __builtin_amdgcn_s_wait_asynccnt(0);
#endif
        __syncthreads();

        // ---- scores: S = (Q K^T) * 1/sqrt(HD) + mask ----
        v8f s_acc[4] = {zero8, zero8, zero8, zero8};
        #pragma unroll
        for (int nt = 0; nt < 4; ++nt) {
            #pragma unroll
            for (int ks = 0; ks < 2; ++ks) {
                FragB16 kf;
                int idx = (nt * 16 + ln) * K2_STR + ks * 32 + half * 8;
                kf.q[0] = *(const uint4*)(k_lds + idx);
                kf.q[1] = *(const uint4*)(k_lds + idx + 16);
                s_acc[nt] = wmma_bf16(qf[ks], kf, s_acc[nt]);
            }
        }
        float mv[4];
        #pragma unroll
        for (int nt = 0; nt < 4; ++nt) mv[nt] = mk[k0 + nt * 16 + ln];
        #pragma unroll
        for (int nt = 0; nt < 4; ++nt)
            #pragma unroll
            for (int r = 0; r < 8; ++r)
                s_acc[nt][r] = s_acc[nt][r] * 0.125f + mv[nt];

        // ---- online softmax (rows live in 16-lane halves of the C layout) ----
        #pragma unroll
        for (int r = 0; r < 8; ++r) {
            float tmax = fmaxf(fmaxf(s_acc[0][r], s_acc[1][r]),
                               fmaxf(s_acc[2][r], s_acc[3][r]));
            #pragma unroll
            for (int i = 1; i < 16; i <<= 1)
                tmax = fmaxf(tmax, __shfl_xor(tmax, i, 32));
            float nm = fmaxf(m_run[r], tmax);
            float sc = __expf(m_run[r] - nm);
            float part = 0.f;
            int row = r + half * 8;
            #pragma unroll
            for (int nt = 0; nt < 4; ++nt) {
                float p = __expf(s_acc[nt][r] - nm);
                part += p;
                pw[row * P_STR + nt * 16 + ln] = bf1(p);
            }
            #pragma unroll
            for (int i = 1; i < 16; i <<= 1)
                part += __shfl_xor(part, i, 32);
            l_run[r] = l_run[r] * sc + part;
            m_run[r] = nm;
            #pragma unroll
            for (int dt = 0; dt < 4; ++dt) o_acc[dt][r] *= sc;
        }

        // ---- O += P V  (P from per-wave LDS slab, V^T tile from LDS) ----
        #pragma unroll
        for (int kk = 0; kk < 64; kk += 32) {
            FragB16 pf;
            int idx = ln * P_STR + kk + half * 8;
            pf.q[0] = *(const uint4*)(pw + idx);
            pf.q[1] = *(const uint4*)(pw + idx + 16);
            #pragma unroll
            for (int dt = 0; dt < 4; ++dt) {
                FragB16 vf;
                int vidx = (dt * 16 + ln) * K2_STR + kk + half * 8;
                vf.q[0] = *(const uint4*)(v_lds + vidx);
                vf.q[1] = *(const uint4*)(v_lds + vidx + 16);
                o_acc[dt] = wmma_bf16(pf, vf, o_acc[dt]);
            }
        }
    }

    // ---- normalize and write ctx in [B, S, NH*HD] f32 ----
    #pragma unroll
    for (int r = 0; r < 8; ++r) {
        float inv = 1.0f / l_run[r];
        int row = q0 + r + half * 8;
        #pragma unroll
        for (int dt = 0; dt < 4; ++dt)
            out[(b * SS + row) * HSZ + head * HD + dt * 16 + ln] = o_acc[dt][r] * inv;
    }
}

// ---------------------------------------------------------------------------
extern "C" void kernel_launch(void* const* d_in, const int* in_sizes, int n_in,
                              void* d_out, int out_size, void* d_ws, size_t ws_size,
                              hipStream_t stream) {
    const float* hs   = (const float*)d_in[0];
    const float* mask = (const float*)d_in[1];
    const float* Wq   = (const float*)d_in[2];
    const float* bq   = (const float*)d_in[3];
    const float* Wk   = (const float*)d_in[4];
    const float* bk   = (const float*)d_in[5];
    const float* Wv   = (const float*)d_in[6];
    const float* bv   = (const float*)d_in[7];
    float* out = (float*)d_out;

    // workspace: Q, K (head-major bf16) and V transposed (bf16), 8 MB each
    const size_t elems = (size_t)MROWS * HSZ;   // 4096*1024
    u16* Qb = (u16*)d_ws;
    u16* Kb = Qb + elems;
    u16* Vb = Kb + elems;

    dim3 g1(MROWS / 128, HSZ / 64, 3), b1(256);
    qkv_gemm_kernel<<<g1, b1, 0, stream>>>(hs, Wq, bq, Wk, bk, Wv, bv, Qb, Kb, Vb);

    dim3 g2(SS / 128, NH, BB), b2(256);
    flash_attn_kernel<<<g2, b2, 0, stream>>>(Qb, Kb, Vb, mask, out);
}
